// ConstituencyAwareAttention_39685497815698
// MI455X (gfx1250) — matrix operand order repaired and checked by reading, not compiled, not run on hardware
//
#include <hip/hip_runtime.h>
#include <hip/hip_bf16.h>

// ---------------- types / helpers ----------------
typedef _Float16 h8_t  __attribute__((ext_vector_type(8)));
typedef _Float16 h16_t __attribute__((ext_vector_type(16)));
typedef float    f8_t  __attribute__((ext_vector_type(8)));
typedef float    f4_t  __attribute__((ext_vector_type(4)));

#define S_LEN   1024
#define HID     1024
#define NHEADS  16
#define HD      64
#define BATCH   8
#define PENALTY 0.5f
#define LOG2E   1.4426950408889634f

static __device__ __forceinline__ h8_t load_h8(const _Float16* p) {
    return *(const h8_t*)p;
}

static __device__ __forceinline__ h8_t cvt_f32x8(const float* p) {
    f4_t a = *(const f4_t*)p;
    f4_t b = *(const f4_t*)(p + 4);
    h8_t r;
    r[0]=(_Float16)a[0]; r[1]=(_Float16)a[1]; r[2]=(_Float16)a[2]; r[3]=(_Float16)a[3];
    r[4]=(_Float16)b[0]; r[5]=(_Float16)b[1]; r[6]=(_Float16)b[2]; r[7]=(_Float16)b[3];
    return r;
}

static __device__ __forceinline__ h16_t cat16(h8_t lo, h8_t hi) {
    return __builtin_shufflevector(lo, hi, 0,1,2,3,4,5,6,7,8,9,10,11,12,13,14,15);
}

static __device__ __forceinline__ f8_t wmma_f16(h16_t a, h16_t b, f8_t c) {
    return __builtin_amdgcn_wmma_f32_16x16x32_f16(false, a, false, b, (short)0, c, false, false);
}

// ---------------- kernel 1: weight transpose + f32->f16 ----------------
// Wt layout: [3][N=1024][K=1024] f16, Wt[m][n][k] = W_m[k][n]
__global__ void caa_wt_kernel(const float* __restrict__ Wq,
                              const float* __restrict__ Wk,
                              const float* __restrict__ Wv,
                              _Float16* __restrict__ Wt) {
    const int mat = blockIdx.y;
    const float* W = (mat == 0) ? Wq : ((mat == 1) ? Wk : Wv);
    const int idx = blockIdx.x * blockDim.x + threadIdx.x;   // 0 .. 1024*1024-1
    const int n = idx >> 10;
    const int k = idx & 1023;
    Wt[(size_t)mat * (HID * HID) + (size_t)n * HID + k] = (_Float16)W[(size_t)k * HID + n];
}

// ---------------- kernel 2: fused QKV projection GEMM ----------------
// One wave computes the SAME 16(M) x 64(N) tile for Q, K and V. Per 32-K step:
// all 24 B-fragment loads are issued first into 12 distinct fragment registers,
// then 12 WMMAs consume them, so older WMMAs overlap younger in-flight loads.
// The A-fragment (f32->f16 converted X rows) is double-buffered across steps.
// Q,K -> [b][h][s][d] f16 ; V -> [b][h][d][s] f16 (transposed for the PV stage)
__global__ void caa_qkv_kernel(const float* __restrict__ X,       // [8192][1024] f32
                               const _Float16* __restrict__ Wt,   // [3][N][K] f16
                               const float* __restrict__ bq,
                               const float* __restrict__ bk,
                               const float* __restrict__ bv,
                               _Float16* __restrict__ Qh,
                               _Float16* __restrict__ Kh,
                               _Float16* __restrict__ Vt) {
    const int wave = threadIdx.x >> 5;
    const int lane = threadIdx.x & 31;
    const int lr   = lane & 15;
    const int g    = lane >> 4;

    const int strip = blockIdx.x * 4 + wave;      // 0..8191
    const int m0 = (strip >> 4) << 4;             // 16-row tile
    const int n0 = (strip & 15) << 6;             // 64-col tile

    const _Float16* W0 = Wt;
    const _Float16* W1 = Wt + (size_t)(HID * HID);
    const _Float16* W2 = Wt + (size_t)2 * (HID * HID);

    const float* xrow = X + (size_t)(m0 + lr) * HID;   // A: lane = M row

    f8_t accQ[4] = {f8_t{}, f8_t{}, f8_t{}, f8_t{}};
    f8_t accK[4] = {f8_t{}, f8_t{}, f8_t{}, f8_t{}};
    f8_t accV[4] = {f8_t{}, f8_t{}, f8_t{}, f8_t{}};

    // A 16x32 f16 fragment: K = g*16 + e, contiguous (matches B)
    h16_t a_cur = cat16(cvt_f32x8(xrow + g * 16), cvt_f32x8(xrow + g * 16 + 8));

    for (int k = 0; k < HID; k += 32) {
        // issue ALL B loads of this step first (12 fragments, 24 x b128)
        h16_t bf[12];
#pragma unroll
        for (int nt = 0; nt < 4; ++nt) {
            // B 32x16 fragment: lane = column n, K = g*16 + e, contiguous in Wt row
            const size_t ro = (size_t)(n0 + nt * 16 + lr) * HID + k + g * 16;
            bf[3 * nt + 0] = cat16(load_h8(W0 + ro), load_h8(W0 + ro + 8));
            bf[3 * nt + 1] = cat16(load_h8(W1 + ro), load_h8(W1 + ro + 8));
            bf[3 * nt + 2] = cat16(load_h8(W2 + ro), load_h8(W2 + ro + 8));
        }
        // prefetch next step's A fragment before consuming current
        h16_t a_nxt = a_cur;
        if (k + 32 < HID)
            a_nxt = cat16(cvt_f32x8(xrow + k + 32 + g * 16),
                          cvt_f32x8(xrow + k + 40 + g * 16));
        // 12 WMMAs; each waits only for its own loads while younger loads fly
#pragma unroll
        for (int nt = 0; nt < 4; ++nt) {
            accQ[nt] = wmma_f16(a_cur, bf[3 * nt + 0], accQ[nt]);
            accK[nt] = wmma_f16(a_cur, bf[3 * nt + 1], accK[nt]);
            accV[nt] = wmma_f16(a_cur, bf[3 * nt + 2], accV[nt]);
        }
        a_cur = a_nxt;
    }

    const int b = m0 >> 10;                        // 1024 rows per batch
#pragma unroll
    for (int nt = 0; nt < 4; ++nt) {
        const int n = n0 + nt * 16 + lr;           // lane's output column
        const int h = n >> 6;
        const int d = n & 63;
        const float bvq = bq[n], bvk = bk[n], bvv = bv[n];
#pragma unroll
        for (int r = 0; r < 8; ++r) {
            const int m = m0 + r + 8 * g;          // C layout: M = r + 8*g
            const int s = m & 1023;
            const size_t qk_idx = (((size_t)b * NHEADS + h) * S_LEN + s) * HD + d;
            Qh[qk_idx] = (_Float16)(accQ[nt][r] + bvq);
            Kh[qk_idx] = (_Float16)(accK[nt][r] + bvk);
            Vt[(((size_t)b * NHEADS + h) * HD + d) * S_LEN + s] =
                (_Float16)(accV[nt][r] + bvv);
        }
    }
}

// ---------------- kernel 3: flash attention with constituency penalty ----------------
// One wave per (b, h, 16-query tile). 32-key steps; penalty is uniform per step
// because q-tiles (16-aligned) and k-steps (32-aligned) never straddle a 64 block.
// K fragments are double-buffered: next-step loads issue before the softmax/LDS
// transpose of the current step.
__global__ void caa_attn_kernel(const _Float16* __restrict__ Qh,
                                const _Float16* __restrict__ Kh,
                                const _Float16* __restrict__ Vt,
                                float* __restrict__ out) {
    __shared__ _Float16 pLds[4][16 * 32];          // per-wave P-transpose scratch (1 KB each)

    const int wave = threadIdx.x >> 5;
    const int lane = threadIdx.x & 31;
    const int lr   = lane & 15;
    const int g    = lane >> 4;

    const int wg = blockIdx.x * 4 + wave;          // 0..8191
    const int b  = wg >> 10;
    const int h  = (wg >> 6) & 15;
    const int q0 = (wg & 63) << 4;

    const _Float16* Qb = Qh + ((size_t)b * NHEADS + h) * (size_t)(S_LEN * HD);
    const _Float16* Kb = Kh + ((size_t)b * NHEADS + h) * (size_t)(S_LEN * HD);
    const _Float16* Vb = Vt + ((size_t)b * NHEADS + h) * (size_t)(HD * S_LEN);

    // Q fragments for the whole tile (A layout, two 32-K slices over hd=64)
    const _Float16* qrow = Qb + (size_t)(q0 + lr) * HD;
    const h16_t a_lo = cat16(load_h8(qrow + g * 16),      load_h8(qrow + g * 16 + 8));
    const h16_t a_hi = cat16(load_h8(qrow + 32 + g * 16), load_h8(qrow + 32 + g * 16 + 8));

    f8_t o[4] = {f8_t{}, f8_t{}, f8_t{}, f8_t{}};
    float mrow[8], lsum[8];
#pragma unroll
    for (int r = 0; r < 8; ++r) { mrow[r] = -3.0e38f; lsum[r] = 0.0f; }

    const float scale = 0.125f;                    // 1/sqrt(64)
    const int   qblk  = q0 >> 6;
    _Float16*   lp    = pLds[wave];

    // B fragments for one 32-key score step: [0]/[1] = cols kb..kb+15 (d slices
    // 0..31 / 32..63), [2]/[3] = cols kb+16..kb+31.
    auto load_kfrags = [&](int kb, h16_t* f) {
        const _Float16* kr0 = Kb + (size_t)(kb + lr) * HD;
        const _Float16* kr1 = Kb + (size_t)(kb + 16 + lr) * HD;
        f[0] = cat16(load_h8(kr0 + g * 16),      load_h8(kr0 + g * 16 + 8));
        f[1] = cat16(load_h8(kr0 + 32 + g * 16), load_h8(kr0 + 32 + g * 16 + 8));
        f[2] = cat16(load_h8(kr1 + g * 16),      load_h8(kr1 + g * 16 + 8));
        f[3] = cat16(load_h8(kr1 + 32 + g * 16), load_h8(kr1 + 32 + g * 16 + 8));
    };

    h16_t kc[4];
    load_kfrags(0, kc);

    for (int kb = 0; kb < S_LEN; kb += 32) {
        // prefetch next step's K fragments (issued before the memory-clobbered wait)
        h16_t kn[4] = {kc[0], kc[1], kc[2], kc[3]};
        if (kb + 32 < S_LEN) load_kfrags(kb + 32, kn);

        // ---- scores: two 16x16 tiles, contraction over hd=64 (2 WMMAs each)
        f8_t s0 = {}; s0 = wmma_f16(a_lo, kc[0], s0); s0 = wmma_f16(a_hi, kc[1], s0);
        f8_t s1 = {}; s1 = wmma_f16(a_lo, kc[2], s1); s1 = wmma_f16(a_hi, kc[3], s1);

        // penalty uniform per (q-tile, 32-key step); fold scale+penalty into log2 units
        const float pen   = (qblk != (kb >> 6)) ? PENALTY : 0.0f;
        const float sc2   = scale * LOG2E;
        const float pen2  = pen * LOG2E;

        // ---- online softmax (rows live across 16-lane halves; xor-butterfly)
        float alpha[8];
#pragma unroll
        for (int r = 0; r < 8; ++r) {
            float v0 = s0[r] * sc2 - pen2;          // log2-domain score
            float v1 = s1[r] * sc2 - pen2;
            s0[r] = v0; s1[r] = v1;
            float mx = fmaxf(v0, v1);
            mx = fmaxf(mx, __shfl_xor(mx, 1));
            mx = fmaxf(mx, __shfl_xor(mx, 2));
            mx = fmaxf(mx, __shfl_xor(mx, 4));
            mx = fmaxf(mx, __shfl_xor(mx, 8));
            float mnew = fmaxf(mrow[r], mx);
            alpha[r] = exp2f(mrow[r] - mnew);
            mrow[r] = mnew;
            float p0 = exp2f(v0 - mnew);
            float p1 = exp2f(v1 - mnew);
            s0[r] = p0; s1[r] = p1;
            float rs = p0 + p1;
            rs += __shfl_xor(rs, 1);
            rs += __shfl_xor(rs, 2);
            rs += __shfl_xor(rs, 4);
            rs += __shfl_xor(rs, 8);
            lsum[r] = lsum[r] * alpha[r] + rs;
        }
#pragma unroll
        for (int nt = 0; nt < 4; ++nt)
#pragma unroll
            for (int r = 0; r < 8; ++r) o[nt][r] *= alpha[r];

        // ---- P: C-layout -> A-layout transpose through per-wave LDS
#pragma unroll
        for (int r = 0; r < 8; ++r) {
            lp[(r + 8 * g) * 32 + lr]      = (_Float16)s0[r];
            lp[(r + 8 * g) * 32 + 16 + lr] = (_Float16)s1[r];
        }
        asm volatile("s_wait_dscnt 0" ::: "memory");
        const _Float16* prow = lp + lr * 32;
        h16_t pa = cat16(*(const h8_t*)(prow + g * 16),
                         *(const h8_t*)(prow + g * 16 + 8));

        // ---- PV: V transposed [d][s] => B fragment is contiguous in key index
#pragma unroll
        for (int nt = 0; nt < 4; ++nt) {
            const _Float16* vrow = Vb + (size_t)(nt * 16 + lr) * S_LEN + kb + g * 16;
            h16_t bvf = cat16(load_h8(vrow), load_h8(vrow + 8));
            o[nt] = wmma_f16(pa, bvf, o[nt]);
        }

        kc[0] = kn[0]; kc[1] = kn[1]; kc[2] = kn[2]; kc[3] = kn[3];
    }

    // ---- finalize and write [b][s][h*64+d] f32
#pragma unroll
    for (int r = 0; r < 8; ++r) lsum[r] = 1.0f / lsum[r];
    float* ob = out + (size_t)b * S_LEN * HID;
#pragma unroll
    for (int nt = 0; nt < 4; ++nt) {
        const int d = h * HD + nt * 16 + lr;
#pragma unroll
        for (int r = 0; r < 8; ++r) {
            const int s = q0 + r + 8 * g;
            ob[(size_t)s * HID + d] = o[nt][r] * lsum[r];
        }
    }
}

// ---------------- launch ----------------
extern "C" void kernel_launch(void* const* d_in, const int* in_sizes, int n_in,
                              void* d_out, int out_size, void* d_ws, size_t ws_size,
                              hipStream_t stream) {
    (void)in_sizes; (void)n_in; (void)out_size; (void)ws_size;

    const float* X  = (const float*)d_in[0];
    const float* Wq = (const float*)d_in[1];
    const float* bq = (const float*)d_in[2];
    const float* Wk = (const float*)d_in[3];
    const float* bk = (const float*)d_in[4];
    const float* Wv = (const float*)d_in[5];
    const float* bv = (const float*)d_in[6];
    float* out = (float*)d_out;

    // workspace layout (f16 elements): Wt[3][1024][1024] | Qh | Kh | Vt (16 MB each)
    _Float16* ws = (_Float16*)d_ws;
    _Float16* Wt = ws;
    _Float16* Qh = Wt + (size_t)3 * HID * HID;
    _Float16* Kh = Qh + (size_t)BATCH * NHEADS * S_LEN * HD;
    _Float16* Vt = Kh + (size_t)BATCH * NHEADS * S_LEN * HD;

    // 1) weight transpose+convert: 1M elements per matrix, 3 matrices
    caa_wt_kernel<<<dim3((HID * HID) / 256, 3), dim3(256), 0, stream>>>(Wq, Wk, Wv, Wt);

    // 2) fused QKV GEMM: 8192 strips of 16x64, 4 waves (128 threads) per block
    caa_qkv_kernel<<<dim3(2048), dim3(128), 0, stream>>>(X, Wt, bq, bk, bv, Qh, Kh, Vt);

    // 3) attention: 8192 (b,h,q-tile) waves, 4 per block
    caa_attn_kernel<<<dim3(2048), dim3(128), 0, stream>>>(Qh, Kh, Vt, out);
}
